// Encoder_7516192768311
// MI455X (gfx1250) — compile-verified, matrix operand order checked by reading, not run on hardware
//
#include <hip/hip_runtime.h>

// ---------------------------------------------------------------------------
// MI455X (gfx1250) fused APL-MLP. Each APL layer == GEMM with K = 3*I via
// hat-basis expansion; bf16 A/B with f32 accumulation using
// v_wmma_f32_16x16x32_bf16. Global->LDS staging uses the gfx1250 async-copy
// path (GLOBAL_LOAD_ASYNC_TO_LDS_*, ASYNCcnt) with double-buffered tiles so
// HBM/L2 latency overlaps the WMMA stream. LayerNorm+ELU+re-expansion fused
// into the GEMM epilogue (full N=512 per block so row stats are local).
// ---------------------------------------------------------------------------

typedef float  v8f   __attribute__((ext_vector_type(8)));
typedef __bf16 v16bf __attribute__((ext_vector_type(16)));

// exact pointee types expected by the async-to-LDS builtins (GCC vector types)
typedef int gv4i __attribute__((vector_size(16)));
typedef int gv2i __attribute__((vector_size(8)));

#if __has_builtin(__builtin_amdgcn_global_load_async_to_lds_b128) && \
    __has_builtin(__builtin_amdgcn_global_load_async_to_lds_b64)
#define HAVE_ASYNC_LDS 1
#else
#define HAVE_ASYNC_LDS 0
#endif

__device__ __forceinline__ void wait_async0() {
#if __has_builtin(__builtin_amdgcn_s_wait_asynccnt)
  __builtin_amdgcn_s_wait_asynccnt(0);
#else
  asm volatile("s_wait_asynccnt 0x0" ::: "memory");
#endif
}

#if HAVE_ASYNC_LDS
__device__ __forceinline__ void async_cp128(const unsigned short* g, unsigned short* l) {
  __builtin_amdgcn_global_load_async_to_lds_b128(
      (__attribute__((address_space(1))) gv4i*)g,
      (__attribute__((address_space(3))) gv4i*)l, 0, 0);
}
__device__ __forceinline__ void async_cp64(const unsigned short* g, unsigned short* l) {
  __builtin_amdgcn_global_load_async_to_lds_b64(
      (__attribute__((address_space(1))) gv2i*)g,
      (__attribute__((address_space(3))) gv2i*)l, 0, 0);
}
#endif

__device__ __forceinline__ unsigned short f2bf(float f) {
  return __builtin_bit_cast(unsigned short, (__bf16)f);
}

// hat basis on uniform knots {0,1,2} of t = clip(x+1, 0, 2)
__device__ __forceinline__ void hat3(float x, float& b0, float& b1, float& b2) {
  float t = fminf(fmaxf(x + 1.0f, 0.0f), 2.0f);
  b0 = fmaxf(1.0f - t, 0.0f);
  b1 = fmaxf(1.0f - fabsf(t - 1.0f), 0.0f);
  b2 = fmaxf(t - 1.0f, 0.0f);
}

// ---------------------------------------------------------------------------
// Weight prep: vals[I][O][3] f32  ->  Wt[O][3I] bf16 (K-contiguous rows)
// ---------------------------------------------------------------------------
__global__ void prep_weights_kernel(const float* __restrict__ W1,
                                    const float* __restrict__ W2,
                                    const float* __restrict__ W3,
                                    unsigned short* __restrict__ W1t,
                                    unsigned short* __restrict__ W2t,
                                    unsigned short* __restrict__ W3t) {
  int e = blockIdx.x * 256 + threadIdx.x;
  const int S1 = 512 * 2304, S2 = 512 * 1536;
  if (e < S1) {
    int o = e / 2304, k = e % 2304, i = k / 3, p = k % 3;
    W1t[e] = f2bf(W1[((size_t)i * 512 + o) * 3 + p]);
  } else if (e < S1 + S2) {
    int n = e - S1, o = n / 1536, k = n % 1536, i = k / 3, p = k % 3;
    W2t[n] = f2bf(W2[((size_t)i * 512 + o) * 3 + p]);
  } else if (e < S1 + 2 * S2) {
    int n = e - S1 - S2, o = n / 1536, k = n % 1536, i = k / 3, p = k % 3;
    W3t[n] = f2bf(W3[((size_t)i * 512 + o) * 3 + p]);
  }
}

// ---------------------------------------------------------------------------
// Tiny K=24 layer (za = elu(apl(action, W_za))) + hat expansion of
// concat(zs, za) into bf16 X1[B][2304]. One block per 16 rows.
// ---------------------------------------------------------------------------
__global__ __launch_bounds__(256)
void za_expand_kernel(const float* __restrict__ zs,
                      const float* __restrict__ action,
                      const float* __restrict__ Wza,
                      unsigned short* __restrict__ X1) {
  __shared__ float w_lds[8 * 256 * 3];   // 24 KB
  __shared__ float za_lds[256];
  const int t = threadIdx.x;
  #pragma unroll
  for (int k = 0; k < 24; ++k) w_lds[k * 256 + t] = Wza[k * 256 + t];
  __syncthreads();

  const int row0 = blockIdx.x * 16;
  for (int r = 0; r < 16; ++r) {
    const int row = row0 + r;
    float acc = 0.f;
    #pragma unroll
    for (int i = 0; i < 8; ++i) {
      float a = action[(size_t)row * 8 + i];
      float b0, b1, b2; hat3(a, b0, b1, b2);
      const float* w = &w_lds[(i * 256 + t) * 3];
      acc += b0 * w[0] + b1 * w[1] + b2 * w[2];
    }
    za_lds[t] = acc > 0.f ? acc : (expf(acc) - 1.f);   // ELU
    __syncthreads();
    unsigned short* orow = X1 + (size_t)row * 2304;
    #pragma unroll
    for (int cc = 0; cc < 3; ++cc) {
      int c = cc * 256 + t;
      float v = (c < 512) ? zs[(size_t)row * 512 + c] : za_lds[c - 512];
      float b0, b1, b2; hat3(v, b0, b1, b2);
      orow[3 * c + 0] = f2bf(b0);
      orow[3 * c + 1] = f2bf(b1);
      orow[3 * c + 2] = f2bf(b2);
    }
    __syncthreads();
  }
}

// ---------------------------------------------------------------------------
// Fused GEMM: C[32 x 512] = A[32 x K] * Wt[512 x K]^T  (bf16 in, f32 acc)
// Block: 256 threads = 8 wave32. Wave w -> N stripe [64w, 64w+64), 2 M-tiles.
// Double-buffered LDS tiles filled with GLOBAL_LOAD_ASYNC_TO_LDS (ASYNCcnt),
// padded to stride 40 halfwords -> conflict-free ds_load_b128 fragment reads.
// FINAL=false: LN + ELU + hat-expand -> bf16 out[B][1536]
// FINAL=true : raw f32 out[B][512]
// ---------------------------------------------------------------------------
#define KPAD 40
#define B_HW (512 * KPAD)          // 20480 halfwords per B buffer (40960 B)
#define A_HW (32 * KPAD)           // 1280 halfwords per A buffer  (2560 B)
#define B_BYTES (B_HW * 2)
#define A_BYTES (A_HW * 2)

template <int K, bool FINAL>
__global__ __launch_bounds__(256)
void apl_gemm_kernel(const unsigned short* __restrict__ A,
                     const unsigned short* __restrict__ W,
                     void* __restrict__ outv) {
  extern __shared__ char smem[];
  // buffer base pointers computed by offset arithmetic (no pointer arrays --
  // those constant-fold into addrspacecast static initializers that lld
  // cannot relocate)
  unsigned short* const lds_hw = (unsigned short*)smem;
  float* const c_lds = (float*)smem;   // [32][520], aliased after the K-loop

  const int tid   = threadIdx.x;
  const int lane  = tid & 31;
  const int wv    = tid >> 5;
  const int l15   = lane & 15;
  const int koff  = (lane >= 16) ? 8 : 0;   // K sub-block per ISA A/B layout
  const int mhalf = (lane >= 16) ? 8 : 0;   // C/D row offset for upper lanes
  const int rowBase = blockIdx.x * 32;
  const int arow = tid >> 3, aseg = tid & 7;

  // Stage one 32-K tile (A: 32x64B via b64/lane, B: 512x64B via 8 b128/lane)
  auto issue = [&](int buf, int kb) {
    unsigned short* bb = lds_hw + buf * B_HW;            // B tile buffer
    unsigned short* ab = lds_hw + 2 * B_HW + buf * A_HW; // A tile buffer
#if HAVE_ASYNC_LDS
    async_cp64(A + (size_t)(rowBase + arow) * K + kb + aseg * 4,
               ab + arow * KPAD + aseg * 4);
    #pragma unroll
    for (int i = 0; i < 8; ++i) {
      int c = i * 256 + tid, row = c >> 2, seg = c & 3;
      async_cp128(W + (size_t)row * K + kb + seg * 8,
                  bb + row * KPAD + seg * 8);
    }
#else
    // fallback: batch all global loads into regs, then all LDS stores
    uint2 ta = *(const uint2*)(A + (size_t)(rowBase + arow) * K + kb + aseg * 4);
    uint4 tb[8];
    #pragma unroll
    for (int i = 0; i < 8; ++i) {
      int c = i * 256 + tid, row = c >> 2, seg = c & 3;
      tb[i] = *(const uint4*)(W + (size_t)row * K + kb + seg * 8);
    }
    *(uint2*)(ab + arow * KPAD + aseg * 4) = ta;
    #pragma unroll
    for (int i = 0; i < 8; ++i) {
      int c = i * 256 + tid, row = c >> 2, seg = c & 3;
      *(uint4*)(bb + row * KPAD + seg * 8) = tb[i];
    }
#endif
  };

  v8f acc[2][4];
  #pragma unroll
  for (int h = 0; h < 2; ++h)
    #pragma unroll
    for (int j = 0; j < 4; ++j)
      acc[h][j] = (v8f){0.f, 0.f, 0.f, 0.f, 0.f, 0.f, 0.f, 0.f};

  issue(0, 0);   // prologue: fill buffer 0

  for (int kb = 0; kb < K; kb += 32) {
    const int cur = (kb >> 5) & 1;
    unsigned short* bb = lds_hw + cur * B_HW;
    unsigned short* ab = lds_hw + 2 * B_HW + cur * A_HW;
    wait_async0();        // my async copies for buf[cur] are done (in-order)
    __syncthreads();      // everyone's copies visible; prev compute reads done
    if (kb + 32 < K)
      issue(cur ^ 1, kb + 32);   // next tile in flight during this compute

    // A fragments: lane l15 = M row; VGPRs 0-3 hold K=koff..koff+7,
    // VGPRs 4-7 hold K=koff+16..koff+23 -> two ds_load_b128 per frag.
    v16bf af[2], bfr[4];
    #pragma unroll
    for (int h = 0; h < 2; ++h) {
      const unsigned short* p = ab + (h * 16 + l15) * KPAD + koff;
      union { uint4 u[2]; v16bf v; } u;
      u.u[0] = *(const uint4*)p;
      u.u[1] = *(const uint4*)(p + 16);
      af[h] = u.v;
    }
    // B fragments (lane = N, K packed identically; Wt rows are K-contiguous)
    #pragma unroll
    for (int j = 0; j < 4; ++j) {
      const unsigned short* p = bb + (wv * 64 + j * 16 + l15) * KPAD + koff;
      union { uint4 u[2]; v16bf v; } u;
      u.u[0] = *(const uint4*)p;
      u.u[1] = *(const uint4*)(p + 16);
      bfr[j] = u.v;
    }
    #pragma unroll
    for (int h = 0; h < 2; ++h)
      #pragma unroll
      for (int j = 0; j < 4; ++j)
        acc[h][j] = __builtin_amdgcn_wmma_f32_16x16x32_bf16(
            false, af[h], false, bfr[j], (short)0, acc[h][j], false, false);
  }

  if (FINAL) {
    // last layer: raw f32, straight from accumulators
    float* outf = (float*)outv;
    #pragma unroll
    for (int h = 0; h < 2; ++h)
      #pragma unroll
      for (int j = 0; j < 4; ++j) {
        int n = wv * 64 + j * 16 + l15;
        int m = h * 16 + mhalf;
        #pragma unroll
        for (int r = 0; r < 8; ++r)
          outf[(size_t)(rowBase + m + r) * 512 + n] = acc[h][j][r];
      }
  } else {
    __syncthreads();   // all frag reads done before C aliases the tile buffers
    #pragma unroll
    for (int h = 0; h < 2; ++h)
      #pragma unroll
      for (int j = 0; j < 4; ++j) {
        int n = wv * 64 + j * 16 + l15;
        int m = h * 16 + mhalf;
        #pragma unroll
        for (int r = 0; r < 8; ++r)
          c_lds[(m + r) * 520 + n] = acc[h][j][r];
      }
    __syncthreads();

    // LayerNorm + ELU + hat re-expansion; wave wv owns rows 4wv..4wv+3
    unsigned short* outb = (unsigned short*)outv;
    #pragma unroll
    for (int rr = 0; rr < 4; ++rr) {
      int row = wv * 4 + rr;
      float vb[16], s = 0.f, sq = 0.f;
      #pragma unroll
      for (int k = 0; k < 16; ++k) {
        float v = c_lds[row * 520 + k * 32 + lane];
        vb[k] = v; s += v; sq += v * v;
      }
      #pragma unroll
      for (int m = 16; m >= 1; m >>= 1) {
        s  += __shfl_xor(s,  m, 32);
        sq += __shfl_xor(sq, m, 32);
      }
      float mu  = s * (1.f / 512.f);
      float var = sq * (1.f / 512.f) - mu * mu;
      float rs  = rsqrtf(var + 1e-5f);
      #pragma unroll
      for (int k = 0; k < 16; ++k) {
        float v = (vb[k] - mu) * rs;
        float e = v > 0.f ? v : (expf(v) - 1.f);   // ELU
        float b0, b1, b2; hat3(e, b0, b1, b2);
        size_t o = (size_t)(rowBase + row) * 1536 + 3 * (k * 32 + lane);
        outb[o + 0] = f2bf(b0);
        outb[o + 1] = f2bf(b1);
        outb[o + 2] = f2bf(b2);
      }
    }
  }
}

// ---------------------------------------------------------------------------
extern "C" void kernel_launch(void* const* d_in, const int* in_sizes, int n_in,
                              void* d_out, int out_size, void* d_ws, size_t ws_size,
                              hipStream_t stream) {
  const float* zs     = (const float*)d_in[0];   // [B,512]
  const float* action = (const float*)d_in[1];   // [B,8]
  const float* Wza    = (const float*)d_in[2];   // [8,256,3]
  const float* W1     = (const float*)d_in[3];   // [768,512,3]
  const float* W2     = (const float*)d_in[4];   // [512,512,3]
  const float* W3     = (const float*)d_in[5];   // [512,512,3]
  const int B = in_sizes[0] / 512;               // 16384

  // workspace layout (bytes), all 16B aligned for b128 loads
  char* ws = (char*)d_ws;
  unsigned short* W1t = (unsigned short*)(ws + 0);         // [512][2304] bf16
  unsigned short* W2t = (unsigned short*)(ws + 2359296);   // [512][1536]
  unsigned short* W3t = (unsigned short*)(ws + 3932160);   // [512][1536]
  unsigned short* X1  = (unsigned short*)(ws + 5505024);   // [B][2304]
  unsigned short* X2  = X1 + (size_t)B * 2304;             // [B][1536]
  unsigned short* X3  = X1;                                 // reuse X1's space

  // LDS: 2 x (40KB B-tile + 2.5KB A-tile) = 87040; C region (66560) aliases
  const size_t LDS_BYTES = 2 * (B_BYTES + A_BYTES);

  {
    int tot = 512 * 2304 + 2 * 512 * 1536;
    prep_weights_kernel<<<(tot + 255) / 256, 256, 0, stream>>>(W1, W2, W3,
                                                               W1t, W2t, W3t);
  }
  za_expand_kernel<<<B / 16, 256, 0, stream>>>(zs, action, Wza, X1);
  apl_gemm_kernel<2304, false><<<B / 32, 256, LDS_BYTES, stream>>>(X1, W1t, (void*)X2);
  apl_gemm_kernel<1536, false><<<B / 32, 256, LDS_BYTES, stream>>>(X2, W2t, (void*)X3);
  apl_gemm_kernel<1536, true ><<<B / 32, 256, LDS_BYTES, stream>>>(X3, W3t, d_out);
}
